// SelectiveSSM_26190710571016
// MI455X (gfx1250) — compile-verified, matrix operand order checked by reading, not run on hardware
//
#include <hip/hip_runtime.h>
#include <hip/hip_bf16.h>
#include <math.h>

#define B_  2
#define L_  2048
#define DM  1024
#define DI  2048
#define DS  16
#define KC  4

typedef __attribute__((ext_vector_type(16))) __bf16 v16bf;
typedef __attribute__((ext_vector_type(8)))  float  v8f;

__device__ __forceinline__ __bf16 f2bf(float f) {
  union { float f; unsigned u; } v; v.f = f;
  unsigned r = (v.u + 0x7FFFu + ((v.u >> 16) & 1u)) >> 16;  // RNE
  union { unsigned short s; __bf16 b; } o; o.s = (unsigned short)r;
  return o.b;
}

// ---------------- elementwise f32 -> bf16 ----------------
__global__ void cvt_f32_bf16(const float* __restrict__ in, __bf16* __restrict__ out, int n) {
  int i = blockIdx.x * blockDim.x + threadIdx.x;
  if (i < n) out[i] = f2bf(in[i]);
}

// ---------- W (KxN f32, row-major) -> Wt (NxK bf16, row-major) ----------
__global__ void transpose_f32_bf16(const float* __restrict__ in, __bf16* __restrict__ out,
                                   int K, int N) {
  int i = blockIdx.x * blockDim.x + threadIdx.x;
  if (i < K * N) {
    int n = i / K, k = i % K;
    out[i] = f2bf(in[(size_t)k * N + n]);
  }
}

// ---------------- bf16 WMMA GEMM: C(MxN,f32) = A(MxK) * Bt(NxK)^T ----------------
// Block tile 128(M) x 128(N) x 32(K); 256 threads = 8 waves; wave tile 32x64 = 8 WMMA.
// Double-buffered LDS filled by async global->LDS DMA (ASYNCcnt), no VGPR round-trip.
__global__ __launch_bounds__(256) void gemm_bf16_wmma(
    const __bf16* __restrict__ A, const __bf16* __restrict__ Bt,
    float* __restrict__ C, int M, int N, int K)
{
  __shared__ __align__(16) __bf16 As[2][128][40];   // 80B rows: 16B-aligned, bank-staggered
  __shared__ __align__(16) __bf16 Bs[2][128][40];

  const int tid  = threadIdx.x;
  const int lane = tid & 31;
  const int wid  = tid >> 5;
  const int wm   = wid & 3;          // 4 wave-rows  (32 M each)
  const int wn   = wid >> 2;         // 2 wave-cols  (64 N each)
  const int m0   = blockIdx.y * 128;
  const int n0   = blockIdx.x * 128;

  const int half = lane >> 4;        // K-half select per ISA 16-bit A/B layout
  const int r    = lane & 15;        // M (A) / N (B) index within tile

  const unsigned aBase = (unsigned)(size_t)&As[0][0][0];  // low 32 bits = LDS offset
  const unsigned bBase = (unsigned)(size_t)&Bs[0][0][0];

  v8f acc[2][4] = {};
  union Frag { uint4 q[2]; v16bf v; };

  const int NT = K >> 5;

  // Issue one K-slab (128x32 A + 128x32 B) via async DMA: 4 x 16B chunks per thread.
  auto issue = [&](int kt, int buf) {
    const int k0 = kt << 5;
    #pragma unroll
    for (int p = 0; p < 4; ++p) {
      int task = p * 256 + tid;                 // 0..1023
      if (p < 2) {                              // A: tasks 0..511
        int row = task >> 2, ch = task & 3;
        unsigned lds = aBase + (unsigned)buf * 10240u + (unsigned)row * 80u + (unsigned)ch * 16u;
        unsigned go  = (unsigned)(((size_t)(m0 + row) * K + k0 + ch * 8) * 2);
        asm volatile("global_load_async_to_lds_b128 %0, %1, %2"
                     :: "v"(lds), "v"(go), "s"(A) : "memory");
      } else {                                  // B: tasks 512..1023
        int t2 = task - 512;
        int row = t2 >> 2, ch = t2 & 3;
        unsigned lds = bBase + (unsigned)buf * 10240u + (unsigned)row * 80u + (unsigned)ch * 16u;
        unsigned go  = (unsigned)(((size_t)(n0 + row) * K + k0 + ch * 8) * 2);
        asm volatile("global_load_async_to_lds_b128 %0, %1, %2"
                     :: "v"(lds), "v"(go), "s"(Bt) : "memory");
      }
    }
  };

  issue(0, 0);                                  // prologue: slab 0 -> buffer 0

  for (int kt = 0; kt < NT; ++kt) {
    const int cur = kt & 1;
    if (kt + 1 < NT) {
      issue(kt + 1, cur ^ 1);                   // prefetch next slab into other buffer
      asm volatile("s_wait_asynccnt 0x4" ::: "memory");  // current slab landed
    } else {
      asm volatile("s_wait_asynccnt 0x0" ::: "memory");
    }
    __syncthreads();

    // Fragments: lanes 0-15 -> K {0..7,16..23}; lanes 16-31 -> K {8..15,24..31}
    Frag a[2], b[4];
    #pragma unroll
    for (int i = 0; i < 2; ++i) {
      a[i].q[0] = *(const uint4*)&As[cur][wm * 32 + i * 16 + r][half * 8];
      a[i].q[1] = *(const uint4*)&As[cur][wm * 32 + i * 16 + r][half * 8 + 16];
    }
    #pragma unroll
    for (int j = 0; j < 4; ++j) {
      b[j].q[0] = *(const uint4*)&Bs[cur][wn * 64 + j * 16 + r][half * 8];
      b[j].q[1] = *(const uint4*)&Bs[cur][wn * 64 + j * 16 + r][half * 8 + 16];
    }
    #pragma unroll
    for (int i = 0; i < 2; ++i)
      #pragma unroll
      for (int j = 0; j < 4; ++j)
        acc[i][j] = __builtin_amdgcn_wmma_f32_16x16x32_bf16(
            false, a[i].v, false, b[j].v, (short)0, acc[i][j], false, false);
    __syncthreads();
  }

  // epilogue: C/D layout: VGPR v -> M = v + 8*half ; N = lane&15
  #pragma unroll
  for (int i = 0; i < 2; ++i)
    #pragma unroll
    for (int j = 0; j < 4; ++j) {
      int mt = m0 + wm * 32 + i * 16 + 8 * half;
      int nt = n0 + wn * 64 + j * 16 + r;
      #pragma unroll
      for (int v = 0; v < 8; ++v)
        C[(size_t)(mt + v) * N + nt] = acc[i][j][v];
    }
}

// ---------------- causal depthwise conv (K=4) + SiLU, dual f32/bf16 output ----------------
__global__ void conv_silu_kernel(const float* __restrict__ xz,
                                 const float* __restrict__ cw,
                                 const float* __restrict__ cb,
                                 float* __restrict__ xcf,
                                 __bf16* __restrict__ xcb)
{
  int i = blockIdx.x * blockDim.x + threadIdx.x;   // over B*L*DI
  if (i >= B_ * L_ * DI) return;
  int c = i % DI;
  int t = (i / DI) % L_;
  int b = i / (DI * L_);
  float acc = cb[c];
  #pragma unroll
  for (int k = 0; k < KC; ++k) {
    int tt = t + k - (KC - 1);
    if (tt >= 0)
      acc += xz[(size_t)(b * L_ + tt) * (2 * DI) + c] * cw[c * KC + k];
  }
  float s = acc / (1.f + expf(-acc));   // SiLU
  xcf[i] = s;
  xcb[i] = f2bf(s);
}

// ---------------- B/C projection: BC[row][0:32] = xc[row,:] @ W_x (2048x32) ----------------
__global__ void bc_kernel(const float* __restrict__ xcf, const float* __restrict__ Wx,
                          float* __restrict__ BC)
{
  int rowid = blockIdx.x;            // 0..B*L-1
  int n = threadIdx.x;               // 0..31
  const float* xr = xcf + (size_t)rowid * DI;
  float acc = 0.f;
  for (int k = 0; k < DI; ++k) acc += xr[k] * Wx[k * (2 * DS) + n];
  BC[rowid * 32 + n] = acc;
}

// ---------------- fused selective scan + softplus + SiLU gate; emits bf16 y ----------------
__global__ __launch_bounds__(256) void scan_kernel(
    const float* __restrict__ dtpre, const float* __restrict__ xcf,
    const float* __restrict__ BC,    const float* __restrict__ A_log,
    const float* __restrict__ Dv,    const float* __restrict__ b_dt,
    const float* __restrict__ xz,    __bf16* __restrict__ ybf)
{
  int bb = blockIdx.x >> 3;                       // batch
  int d  = (blockIdx.x & 7) * 256 + threadIdx.x;  // channel
  __shared__ float sBC[32];                       // [0:16]=B_t, [16:32]=C_t

  float Arow[DS];
  #pragma unroll
  for (int s = 0; s < DS; ++s) Arow[s] = -expf(A_log[d * DS + s]);
  float Dd = Dv[d], bd = b_dt[d];
  float h[DS];
  #pragma unroll
  for (int s = 0; s < DS; ++s) h[s] = 0.f;

  for (int t = 0; t < L_; ++t) {
    size_t rp = (size_t)(bb * L_ + t);
    if (threadIdx.x < 32) sBC[threadIdx.x] = BC[rp * 32 + threadIdx.x];
    __syncthreads();

    float xv = dtpre[rp * DI + d] + bd;
    float dt = xv > 20.f ? xv : log1pf(expf(xv));  // softplus
    float u  = xcf[rp * DI + d];
    float du = dt * u;
    float y  = Dd * u;
    #pragma unroll
    for (int s = 0; s < DS; ++s) {
      h[s] = expf(dt * Arow[s]) * h[s] + du * sBC[s];
      y += h[s] * sBC[DS + s];
    }
    float z = xz[rp * (2 * DI) + DI + d];
    y *= z / (1.f + expf(-z));                     // * SiLU(z)
    ybf[rp * DI + d] = f2bf(y);
    __syncthreads();
  }
}

extern "C" void kernel_launch(void* const* d_in, const int* in_sizes, int n_in,
                              void* d_out, int out_size, void* d_ws, size_t ws_size,
                              hipStream_t stream) {
  (void)in_sizes; (void)n_in; (void)out_size; (void)ws_size;
  const float* x      = (const float*)d_in[0];
  const float* W_in   = (const float*)d_in[1];
  const float* conv_w = (const float*)d_in[2];
  const float* conv_b = (const float*)d_in[3];
  const float* W_x    = (const float*)d_in[4];
  const float* W_dt   = (const float*)d_in[5];
  const float* b_dt   = (const float*)d_in[6];
  const float* A_log  = (const float*)d_in[7];
  const float* Dv     = (const float*)d_in[8];
  const float* W_out  = (const float*)d_in[9];
  float* out = (float*)d_out;

  char* w = (char*)d_ws;
  size_t off = 0;
  auto alloc = [&](size_t bytes) -> void* {
    void* p = w + off; off += (bytes + 255) & ~(size_t)255; return p;
  };
  const int M = B_ * L_;                                   // 4096 token rows
  __bf16* xbf   = (__bf16*)alloc((size_t)M * DM * 2);
  __bf16* WinT  = (__bf16*)alloc((size_t)(2 * DI) * DM * 2);
  float*  xzbuf = (float*) alloc((size_t)M * (2 * DI) * 4);
  float*  xcf   = (float*) alloc((size_t)M * DI * 4);
  __bf16* xcb   = (__bf16*)alloc((size_t)M * DI * 2);
  __bf16* WdtT  = (__bf16*)alloc((size_t)DI * DI * 2);
  float*  dtpre = (float*) alloc((size_t)M * DI * 4);
  float*  BC    = (float*) alloc((size_t)M * 32 * 4);
  __bf16* ybf   = (__bf16*)alloc((size_t)M * DI * 2);
  __bf16* WoutT = (__bf16*)alloc((size_t)DM * DI * 2);

  // precision prep
  cvt_f32_bf16<<<(M * DM + 255) / 256, 256, 0, stream>>>(x, xbf, M * DM);
  transpose_f32_bf16<<<((2 * DI) * DM + 255) / 256, 256, 0, stream>>>(W_in, WinT, DM, 2 * DI);
  transpose_f32_bf16<<<(DI * DI + 255) / 256, 256, 0, stream>>>(W_dt, WdtT, DI, DI);
  transpose_f32_bf16<<<(DM * DI + 255) / 256, 256, 0, stream>>>(W_out, WoutT, DI, DM);

  // GEMM1: xz = x @ W_in   (4096 x 4096 x K=1024)
  gemm_bf16_wmma<<<dim3((2 * DI) / 128, M / 128), 256, 0, stream>>>(xbf, WinT, xzbuf, M, 2 * DI, DM);

  // conv + SiLU
  conv_silu_kernel<<<(M * DI + 255) / 256, 256, 0, stream>>>(xzbuf, conv_w, conv_b, xcf, xcb);

  // GEMM2: dt_pre = xc @ W_dt   (4096 x 2048 x K=2048)
  gemm_bf16_wmma<<<dim3(DI / 128, M / 128), 256, 0, stream>>>(xcb, WdtT, dtpre, M, DI, DI);

  // B/C projections (N=32, memory-trivial)
  bc_kernel<<<M, 32, 0, stream>>>(xcf, W_x, BC);

  // sequential selective scan + gate -> bf16 y
  scan_kernel<<<16, 256, 0, stream>>>(dtpre, xcf, BC, A_log, Dv, b_dt, xzbuf, ybf);

  // GEMM3: out = y @ W_out   (4096 x 1024 x K=2048) -> fp32 output
  gemm_bf16_wmma<<<dim3(DM / 128, M / 128), 256, 0, stream>>>(ybf, WoutT, out, M, DM, DI);
}
